// JointNet_22462678958223
// MI455X (gfx1250) — compile-verified, matrix operand order checked by reading, not run on hardware
//
#include <hip/hip_runtime.h>

// ---- POD vector types (safe in unions, guaranteed b128 loads) ----
typedef __attribute__((ext_vector_type(16))) __bf16 v16bf;
typedef __attribute__((ext_vector_type(8)))  float  v8f;
typedef __attribute__((ext_vector_type(4)))  float  f4v;
typedef __attribute__((ext_vector_type(4)))  unsigned int u4v;

union Frag { u4v u[2]; v16bf v; };

#define B_ 4
#define T_ 256
#define U_ 64
#define A_ 512
#define D_ 640
#define H_ 640
#define V_ 1024

__device__ __forceinline__ float fast_tanh(float x) {
#if __has_builtin(__builtin_amdgcn_tanhf)
  return __builtin_amdgcn_tanhf(x);            // gfx1250 v_tanh_f32
#else
  float e = __builtin_amdgcn_exp2f(-2.8853900817779268f * __builtin_fabsf(x));
  float r = (1.0f - e) * __builtin_amdgcn_rcpf(1.0f + e);
  return __builtin_copysignf(r, x);
#endif
}

// ---------------------------------------------------------------------------
// Pack a (K x N) row-major f32 weight matrix into per-WMMA-fragment bf16
// layout: tile (kc,nt) is 32x16; lane L holds its 16 values contiguously at
// tile*512 + L*16 (elements e: K = kc*32 + e + (e>=8?8:0) + (L>=16?8:0),
// N = nt*16 + (L&15)), matching the 16-bit A/B 16x32 VGPR layout.
// ---------------------------------------------------------------------------
__global__ void pack_w_kernel(const float* __restrict__ W, __bf16* __restrict__ out,
                              int K, int N) {
  int idx = blockIdx.x * 256 + threadIdx.x;
  if (idx >= K * N) return;
  int NT   = N >> 4;
  int tile = idx >> 9;
  int lane = (idx >> 4) & 31;
  int e    = idx & 15;
  int kc   = tile / NT;
  int nt   = tile - kc * NT;
  int n    = (nt << 4) + (lane & 15);
  int k    = (kc << 5) + e + ((e >= 8) ? 8 : 0) + ((lane >= 16) ? 8 : 0);
  out[idx] = (__bf16)W[(size_t)k * N + n];
}

// ---------------------------------------------------------------------------
// Projection GEMM: Out(M,640) = A(M,K)_f32 @ Wpacked(K,640)_bf16 + bias.
// 256 threads = 8 waves; WG tile = 32 rows x 128 cols; each wave owns ONE
// n-tile and TWO m-tiles (B fragment reused by 2 WMMAs). N=640 -> grid.x=5,
// M%32==0 -> no partial tiles. K compile-time -> exact unroll, static LDS.
// ---------------------------------------------------------------------------
template <int K>
__global__ __launch_bounds__(256, 1)
void proj_gemm_kernel(const float* __restrict__ Amat,
                      const __bf16* __restrict__ Wp,
                      const float* __restrict__ bias,
                      float* __restrict__ Out) {
  constexpr int LDK = K + 8;
  __shared__ __align__(16) __bf16 at[32 * LDK];
  const int tid  = threadIdx.x;
  const int lane = tid & 31;
  const int wave = tid >> 5;
  const int hl   = lane >> 4;
  const int r0   = blockIdx.y * 32;

  { // fill 32-row A tile (fp32 -> bf16)
    int ty = tid >> 3, tx = tid & 7;
    const float* arow = Amat + (size_t)(r0 + ty) * K;
    __bf16* lrow = at + ty * LDK;
#pragma unroll
    for (int k0 = tx * 8; k0 < K; k0 += 64) {
      f4v f0 = *reinterpret_cast<const f4v*>(arow + k0);
      f4v f1 = *reinterpret_cast<const f4v*>(arow + k0 + 4);
      union { __bf16 h[8]; u4v u; } p;
      p.h[0]=(__bf16)f0.x; p.h[1]=(__bf16)f0.y; p.h[2]=(__bf16)f0.z; p.h[3]=(__bf16)f0.w;
      p.h[4]=(__bf16)f1.x; p.h[5]=(__bf16)f1.y; p.h[6]=(__bf16)f1.z; p.h[7]=(__bf16)f1.w;
      *reinterpret_cast<u4v*>(lrow + k0) = p.u;
    }
  }
  __syncthreads();

  const int ntg    = blockIdx.x * 8 + wave;         // global n-tile (NT = 40)
  const int m      = lane & 15;
  const int kb_off = hl ? 8 : 0;
  const char* a0p = (const char*)(at + m * LDK + kb_off);
  const char* a1p = (const char*)(at + (16 + m) * LDK + kb_off);
  const char* bp  = (const char*)Wp + ntg * 1024 + lane * 32;

  v8f c0 = {}, c1 = {};
#pragma unroll 4
  for (int kc = 0; kc < K / 32; ++kc) {
    Frag a0, a1, b;
    a0.u[0] = *(const u4v*)(a0p);      a0.u[1] = *(const u4v*)(a0p + 32);
    a1.u[0] = *(const u4v*)(a1p);      a1.u[1] = *(const u4v*)(a1p + 32);
    b.u[0]  = *(const u4v*)(bp);       b.u[1]  = *(const u4v*)(bp + 16);
    a0p += 64; a1p += 64; bp += 40 * 1024;
    c0 = __builtin_amdgcn_wmma_f32_16x16x32_bf16(false, a0.v, false, b.v,
                                                 (short)0, c0, false, false);
    c1 = __builtin_amdgcn_wmma_f32_16x16x32_bf16(false, a1.v, false, b.v,
                                                 (short)0, c1, false, false);
  }

  const int n = blockIdx.x * 128 + wave * 16 + (lane & 15);
  const float bb = bias[n];
  float* o0 = Out + (size_t)(r0 + hl * 8) * H_ + n;
  float* o1 = Out + (size_t)(r0 + 16 + hl * 8) * H_ + n;
#pragma unroll
  for (int r = 0; r < 8; ++r) o0[(size_t)r * H_] = c0[r] + bb;
#pragma unroll
  for (int r = 0; r < 8; ++r) o1[(size_t)r * H_] = c1[r] + bb;
}

// ---------------------------------------------------------------------------
// Fused joint kernel: per WG (b, t, 32 u's):
//   z = tanh(enc_row + dec_rows) -> LDS bf16 (32x640, stride 648)
//   logits = z @ Wout_packed + b_out   (bf16 WMMA, f32 acc; V=1024 over 8 waves)
//   out = log_softmax(logits)          (shfl + LDS reductions), streamed out.
// M-tile = 32: every B fragment feeds TWO WMMAs (halves L2 B-traffic).
// B fragments loaded in two groups of 4 per k-step to bound VGPR pressure.
// ---------------------------------------------------------------------------
__global__ __launch_bounds__(256, 1)
void joint_main_kernel(const float* __restrict__ encp,
                       const float* __restrict__ decp,
                       const __bf16* __restrict__ Wp,
                       const float* __restrict__ bias,
                       float* __restrict__ out) {
  __shared__ __align__(16) __bf16 zt[32 * 648];
  __shared__ float red[8][32];
  __shared__ float rowmax[32];
  __shared__ float rowlog[32];

  const int tid  = threadIdx.x;
  const int lane = tid & 31;
  const int wave = tid >> 5;
  const int hl   = lane >> 4;

  const int u0 = blockIdx.x * 32;
  const int t  = blockIdx.y;
  const int b  = blockIdx.z;
  const int rowbase = (b * T_ + t) * U_ + u0;

  const float* erow = encp + (size_t)(b * T_ + t) * H_;
  { // build tanh(enc+dec) tile: 32 rows x 640
    int ty = tid >> 3, tx = tid & 7;
    const float* drow = decp + (size_t)(b * U_ + u0 + ty) * H_;
    __bf16* zrow = zt + ty * 648;
#pragma unroll
    for (int k0 = tx * 8; k0 < H_; k0 += 64) {
      f4v e0 = *reinterpret_cast<const f4v*>(erow + k0);
      f4v e1 = *reinterpret_cast<const f4v*>(erow + k0 + 4);
      f4v d0 = *reinterpret_cast<const f4v*>(drow + k0);
      f4v d1 = *reinterpret_cast<const f4v*>(drow + k0 + 4);
      union { __bf16 h[8]; u4v u; } p;
      p.h[0] = (__bf16)fast_tanh(e0.x + d0.x);
      p.h[1] = (__bf16)fast_tanh(e0.y + d0.y);
      p.h[2] = (__bf16)fast_tanh(e0.z + d0.z);
      p.h[3] = (__bf16)fast_tanh(e0.w + d0.w);
      p.h[4] = (__bf16)fast_tanh(e1.x + d1.x);
      p.h[5] = (__bf16)fast_tanh(e1.y + d1.y);
      p.h[6] = (__bf16)fast_tanh(e1.z + d1.z);
      p.h[7] = (__bf16)fast_tanh(e1.w + d1.w);
      *reinterpret_cast<u4v*>(zrow + k0) = p.u;
    }
  }
  __syncthreads();

  float biasr[8];
#pragma unroll
  for (int nt = 0; nt < 8; ++nt)
    biasr[nt] = bias[wave * 128 + nt * 16 + (lane & 15)];

  v8f zero = {};
  v8f acc[2][8];
#pragma unroll
  for (int g = 0; g < 2; ++g)
#pragma unroll
    for (int nt = 0; nt < 8; ++nt) acc[g][nt] = zero;

  const int m      = lane & 15;
  const int kb_off = hl ? 8 : 0;
  const char* a0base = (const char*)(zt + m * 648 + kb_off);        // +kc*64 (+32)
  const char* a1base = (const char*)(zt + (16 + m) * 648 + kb_off);
  const char* bbase  = (const char*)Wp + wave * 8192 + lane * 32;   // +kc*65536+nt*1024(+16)

#pragma unroll 1
  for (int kc = 0; kc < H_ / 32; ++kc) {
    Frag a0, a1;
    a0.u[0] = *(const u4v*)(a0base + kc * 64);
    a0.u[1] = *(const u4v*)(a0base + kc * 64 + 32);
    a1.u[0] = *(const u4v*)(a1base + kc * 64);
    a1.u[1] = *(const u4v*)(a1base + kc * 64 + 32);
#pragma unroll
    for (int grp = 0; grp < 2; ++grp) {
      Frag bf[4];
#pragma unroll
      for (int j = 0; j < 4; ++j) {
        int nt = grp * 4 + j;
        bf[j].u[0] = *(const u4v*)(bbase + kc * 65536 + nt * 1024);
        bf[j].u[1] = *(const u4v*)(bbase + kc * 65536 + nt * 1024 + 16);
      }
#pragma unroll
      for (int j = 0; j < 4; ++j) {
        int nt = grp * 4 + j;
        acc[0][nt] = __builtin_amdgcn_wmma_f32_16x16x32_bf16(false, a0.v, false, bf[j].v,
                                                             (short)0, acc[0][nt], false, false);
        acc[1][nt] = __builtin_amdgcn_wmma_f32_16x16x32_bf16(false, a1.v, false, bf[j].v,
                                                             (short)0, acc[1][nt], false, false);
      }
    }
  }

  // ---- fused log-softmax over V=1024 for 32 rows ----
  float pm[2][8];
#pragma unroll
  for (int g = 0; g < 2; ++g)
#pragma unroll
    for (int r = 0; r < 8; ++r) pm[g][r] = -3.4e38f;
#pragma unroll
  for (int g = 0; g < 2; ++g)
#pragma unroll
    for (int nt = 0; nt < 8; ++nt)
#pragma unroll
      for (int r = 0; r < 8; ++r)
        pm[g][r] = fmaxf(pm[g][r], acc[g][nt][r] + biasr[nt]);
  for (int off = 1; off < 16; off <<= 1)
#pragma unroll
    for (int g = 0; g < 2; ++g)
#pragma unroll
      for (int r = 0; r < 8; ++r)
        pm[g][r] = fmaxf(pm[g][r], __shfl_xor(pm[g][r], off, 32));
  if ((lane & 15) == 0)
#pragma unroll
    for (int g = 0; g < 2; ++g)
#pragma unroll
      for (int r = 0; r < 8; ++r) red[wave][g * 16 + hl * 8 + r] = pm[g][r];
  __syncthreads();
  if (tid < 32) {
    float fm = -3.4e38f;
    for (int w = 0; w < 8; ++w) fm = fmaxf(fm, red[w][tid]);
    rowmax[tid] = fm;
  }
  __syncthreads();

  float rm[2][8], ps[2][8];
#pragma unroll
  for (int g = 0; g < 2; ++g)
#pragma unroll
    for (int r = 0; r < 8; ++r) {
      rm[g][r] = rowmax[g * 16 + hl * 8 + r];
      ps[g][r] = 0.0f;
    }
#pragma unroll
  for (int g = 0; g < 2; ++g)
#pragma unroll
    for (int nt = 0; nt < 8; ++nt)
#pragma unroll
      for (int r = 0; r < 8; ++r)
        ps[g][r] += __expf(acc[g][nt][r] + biasr[nt] - rm[g][r]);
  for (int off = 1; off < 16; off <<= 1)
#pragma unroll
    for (int g = 0; g < 2; ++g)
#pragma unroll
      for (int r = 0; r < 8; ++r)
        ps[g][r] += __shfl_xor(ps[g][r], off, 32);
  if ((lane & 15) == 0)
#pragma unroll
    for (int g = 0; g < 2; ++g)
#pragma unroll
      for (int r = 0; r < 8; ++r) red[wave][g * 16 + hl * 8 + r] = ps[g][r];
  __syncthreads();
  if (tid < 32) {
    float s = 0.0f;
    for (int w = 0; w < 8; ++w) s += red[w][tid];
    rowlog[tid] = rowmax[tid] + __logf(s);
  }
  __syncthreads();

  float rl[2][8];
#pragma unroll
  for (int g = 0; g < 2; ++g)
#pragma unroll
    for (int r = 0; r < 8; ++r) rl[g][r] = rowlog[g * 16 + hl * 8 + r];

#pragma unroll
  for (int g = 0; g < 2; ++g)
#pragma unroll
    for (int nt = 0; nt < 8; ++nt) {
      int col = wave * 128 + nt * 16 + (lane & 15);
      float* op = out + (size_t)(rowbase + g * 16 + hl * 8) * V_ + col;
#pragma unroll
      for (int r = 0; r < 8; ++r)
        op[(size_t)r * V_] = acc[g][nt][r] + biasr[nt] - rl[g][r];
    }
}

// ---------------------------------------------------------------------------
extern "C" void kernel_launch(void* const* d_in, const int* in_sizes, int n_in,
                              void* d_out, int out_size, void* d_ws, size_t ws_size,
                              hipStream_t stream) {
  (void)in_sizes; (void)n_in; (void)out_size; (void)ws_size;
  const float* enc_in = (const float*)d_in[0];
  const float* dec_in = (const float*)d_in[1];
  const float* W_enc  = (const float*)d_in[2];
  const float* b_enc  = (const float*)d_in[3];
  const float* W_dec  = (const float*)d_in[4];
  const float* b_dec  = (const float*)d_in[5];
  const float* W_out  = (const float*)d_in[6];
  const float* b_out  = (const float*)d_in[7];
  float* out = (float*)d_out;

  // workspace layout (~6.1 MB total)
  float*  enc_proj = (float*)d_ws;                            // 1024*640 f32
  float*  dec_proj = enc_proj + (size_t)1024 * 640;           // 256*640  f32
  __bf16* wenc_p   = (__bf16*)(dec_proj + (size_t)256 * 640); // 512*640
  __bf16* wdec_p   = wenc_p + (size_t)512 * 640;              // 640*640
  __bf16* wout_p   = wdec_p + (size_t)640 * 640;              // 640*1024

  // pack weights to WMMA fragment layout (L2-resident thereafter)
  pack_w_kernel<<<dim3((A_ * H_) / 256), dim3(256), 0, stream>>>(W_enc, wenc_p, A_, H_);
  pack_w_kernel<<<dim3((D_ * H_) / 256), dim3(256), 0, stream>>>(W_dec, wdec_p, D_, H_);
  pack_w_kernel<<<dim3((H_ * V_) / 256), dim3(256), 0, stream>>>(W_out, wout_p, H_, V_);

  // enc: (1024 x 512) @ (512 x 640);  dec: (256 x 640) @ (640 x 640)
  proj_gemm_kernel<A_><<<dim3(H_ / 128, (B_ * T_) / 32), dim3(256), 0, stream>>>(
      enc_in, wenc_p, b_enc, enc_proj);
  proj_gemm_kernel<D_><<<dim3(H_ / 128, (B_ * U_) / 32), dim3(256), 0, stream>>>(
      dec_in, wdec_p, b_dec, dec_proj);

  // fused tanh + logits GEMM + log-softmax
  joint_main_kernel<<<dim3(U_ / 32, T_, B_), dim3(256), 0, stream>>>(
      enc_proj, dec_proj, wout_p, b_out, out);
}
